// EdgeDecoder_11793980195324
// MI455X (gfx1250) — compile-verified, matrix operand order checked by reading, not run on hardware
//
#include <hip/hip_runtime.h>

typedef __attribute__((ext_vector_type(16))) _Float16 v16h;
typedef __attribute__((ext_vector_type(8)))  _Float16 v8h;
typedef __attribute__((ext_vector_type(8)))  float    v8f;
typedef __attribute__((ext_vector_type(4)))  float    v4f;

#define HID     128
#define WSTRIDE 136   // padded halves/row: row-to-row LDS bank offset = 4 -> conflict-free frag loads

__global__ __launch_bounds__(256)
void edge_mlp_wmma_kernel(const float* __restrict__ x,
                          const int*   __restrict__ src,
                          const int*   __restrict__ dst,
                          const float* __restrict__ W1,
                          const float* __restrict__ b1,
                          const float* __restrict__ W2,
                          const float* __restrict__ b2,
                          const float* __restrict__ W3,
                          const float* __restrict__ b3,
                          float* __restrict__ out,
                          int nEdges, int nTiles)
{
    __shared__ _Float16 W1h[HID * WSTRIDE];
    __shared__ _Float16 W2h[HID * WSTRIDE];
    __shared__ float b1s[HID];
    __shared__ float b2s[HID];
    __shared__ float w3s[HID];

    const int tid = threadIdx.x;

    // ---- stage weights (f32 -> f16, padded) and biases into LDS ----
    for (int i = tid; i < HID * HID; i += blockDim.x) {
        int r = i >> 7, c = i & 127;
        W1h[r * WSTRIDE + c] = (_Float16)W1[i];
        W2h[r * WSTRIDE + c] = (_Float16)W2[i];
    }
    for (int i = tid; i < HID; i += blockDim.x) {
        b1s[i] = b1[i];
        b2s[i] = b2[i];
        w3s[i] = W3[i];
    }
    __syncthreads();

    const int lane  = tid & 31;
    const int col   = lane & 15;               // edge column within tile (also W row within m-tile)
    const int hi    = (lane >> 4) << 3;        // 0 (lanes 0-15) or 8 (lanes 16-31)
    const int wave  = tid >> 5;
    const int wpb   = blockDim.x >> 5;
    const int tile  = blockIdx.x * wpb + wave; // exactly one 16-edge tile per wave: no loop,
                                               // so weight fragments can't be hoisted/spilled
    if (tile >= nTiles) return;                // wave-uniform exit; EXEC stays full for WMMA

    const int e  = tile * 16 + col;
    const int ec = (e < nEdges) ? e : (nEdges - 1);   // clamp keeps EXEC uniform
    const float* __restrict__ xs = x + (size_t)src[ec] * HID;
    const float* __restrict__ xd = x + (size_t)dst[ec] * HID;
    const float bias3 = b3[0];

    // ---- build layer-1 B fragments (edges = N): h0[k][e] = xs[k]*xd[k] ----
    v16h Bf[4];
    #pragma unroll
    for (int kb = 0; kb < 4; ++kb) {
        const int k0 = kb * 32 + hi;       // elements 0..7
        const int k1 = k0 + 16;            // elements 8..15
        v4f a0 = *(const v4f*)(xs + k0);
        v4f a1 = *(const v4f*)(xs + k0 + 4);
        v4f a2 = *(const v4f*)(xs + k1);
        v4f a3 = *(const v4f*)(xs + k1 + 4);
        v4f c0 = *(const v4f*)(xd + k0);
        v4f c1 = *(const v4f*)(xd + k0 + 4);
        v4f c2 = *(const v4f*)(xd + k1);
        v4f c3 = *(const v4f*)(xd + k1 + 4);
        #pragma unroll
        for (int t = 0; t < 4; ++t) {
            Bf[kb][t]      = (_Float16)(a0[t] * c0[t]);
            Bf[kb][4 + t]  = (_Float16)(a1[t] * c1[t]);
            Bf[kb][8 + t]  = (_Float16)(a2[t] * c2[t]);
            Bf[kb][12 + t] = (_Float16)(a3[t] * c3[t]);
        }
    }

    // ---- layer 1: acc = W1 * h0 + b1, relu; repack D tiles into layer-2 B frags ----
    v16h B2[4];
    #pragma unroll
    for (int mb = 0; mb < 8; ++mb) {
        const int mrow = mb * 16;
        v4f bl0 = *(const v4f*)(b1s + mrow + hi);
        v4f bl1 = *(const v4f*)(b1s + mrow + hi + 4);
        v8f acc = {bl0[0], bl0[1], bl0[2], bl0[3], bl1[0], bl1[1], bl1[2], bl1[3]};
        #pragma unroll
        for (int kb = 0; kb < 4; ++kb) {
            const _Float16* wp = &W1h[(mrow + col) * WSTRIDE + kb * 32 + hi];
            v8h lo = *(const v8h*)(wp);
            v8h hg = *(const v8h*)(wp + 16);
            v16h af;
            #pragma unroll
            for (int t = 0; t < 8; ++t) { af[t] = lo[t]; af[8 + t] = hg[t]; }
            acc = __builtin_amdgcn_wmma_f32_16x16x32_f16(
                      false, af, false, Bf[kb], (short)0, acc, false, false);
        }
        #pragma unroll
        for (int t = 0; t < 8; ++t) {
            float v = acc[t] > 0.f ? acc[t] : 0.f;
            // D tile mb feeds B2[mb/2], elements (mb&1)*8 .. +7
            B2[mb >> 1][((mb & 1) << 3) + t] = (_Float16)v;
        }
    }

    // ---- layer 2 + layer 3 fused: acc = relu(W2*h1 + b2); partial += w3 . acc ----
    float partial = 0.f;
    #pragma unroll
    for (int mb = 0; mb < 8; ++mb) {
        const int mrow = mb * 16;
        v4f bl0 = *(const v4f*)(b2s + mrow + hi);
        v4f bl1 = *(const v4f*)(b2s + mrow + hi + 4);
        v8f acc = {bl0[0], bl0[1], bl0[2], bl0[3], bl1[0], bl1[1], bl1[2], bl1[3]};
        #pragma unroll
        for (int kb = 0; kb < 4; ++kb) {
            const _Float16* wp = &W2h[(mrow + col) * WSTRIDE + kb * 32 + hi];
            v8h lo = *(const v8h*)(wp);
            v8h hg = *(const v8h*)(wp + 16);
            v16h af;
            #pragma unroll
            for (int t = 0; t < 8; ++t) { af[t] = lo[t]; af[8 + t] = hg[t]; }
            acc = __builtin_amdgcn_wmma_f32_16x16x32_f16(
                      false, af, false, B2[kb], (short)0, acc, false, false);
        }
        v4f w0 = *(const v4f*)(w3s + mrow + hi);
        v4f w1 = *(const v4f*)(w3s + mrow + hi + 4);
        #pragma unroll
        for (int t = 0; t < 4; ++t) {
            float v0 = acc[t]     > 0.f ? acc[t]     : 0.f;
            float v1 = acc[4 + t] > 0.f ? acc[4 + t] : 0.f;
            partial = fmaf(w0[t], v0, partial);
            partial = fmaf(w1[t], v1, partial);
        }
    }

    // lanes l and l+16 hold the two feature-halves of the same edge
    partial += __shfl_xor(partial, 16, 32);
    if (lane < 16 && e < nEdges) out[e] = partial + bias3;
}

extern "C" void kernel_launch(void* const* d_in, const int* in_sizes, int n_in,
                              void* d_out, int out_size, void* d_ws, size_t ws_size,
                              hipStream_t stream)
{
    const float* x   = (const float*)d_in[0];
    const int*   src = (const int*)  d_in[1];
    const int*   dst = (const int*)  d_in[2];
    const float* W1  = (const float*)d_in[3];
    const float* b1  = (const float*)d_in[4];
    const float* W2  = (const float*)d_in[5];
    const float* b2  = (const float*)d_in[6];
    const float* W3  = (const float*)d_in[7];
    const float* b3  = (const float*)d_in[8];
    float* out = (float*)d_out;

    const int nEdges = in_sizes[1];
    const int nTiles = (nEdges + 15) / 16;

    const int threads = 256;               // 8 wave32s per block, one tile per wave
    const int wpb = threads / 32;
    const int blocks = (nTiles + wpb - 1) / wpb;

    edge_mlp_wmma_kernel<<<blocks, threads, 0, stream>>>(
        x, src, dst, W1, b1, W2, b2, W3, b3, out, nEdges, nTiles);
}
